// AttentionBlock_56573309223733
// MI455X (gfx1250) — matrix operand order repaired and checked by reading, not compiled, not run on hardware
//
#include <hip/hip_runtime.h>
#include <hip/hip_bf16.h>

typedef __attribute__((ext_vector_type(16))) _Float16 v16h;
typedef __attribute__((ext_vector_type(8)))  _Float16 v8h;
typedef __attribute__((ext_vector_type(8)))  float    v8f;

#define B_   16
#define C_   512
#define N_   4096
#define G_   32
#define CG_  16
#define C3_  1536
#define R_   (B_*N_)      // 65536 flat rows (b,n)

// ---------------------------------------------------------------- groupnorm stats
// One block per (b,group): reduce 16*4096 = 65536 contiguous floats.
__global__ void gn_stats_kernel(const float* __restrict__ x, float* __restrict__ stats) {
    __shared__ float s_sum[256], s_sq[256];
    const int bg = blockIdx.x;                       // b*32 + g
    const float* p = x + (size_t)bg * (CG_ * N_);    // (b*C + g*16)*N == bg*65536
    float s = 0.f, q = 0.f;
    for (int i = threadIdx.x; i < CG_ * N_; i += 256) {
        float v = p[i]; s += v; q += v * v;
    }
    s_sum[threadIdx.x] = s; s_sq[threadIdx.x] = q;
    __syncthreads();
    for (int off = 128; off > 0; off >>= 1) {
        if (threadIdx.x < off) {
            s_sum[threadIdx.x] += s_sum[threadIdx.x + off];
            s_sq [threadIdx.x] += s_sq [threadIdx.x + off];
        }
        __syncthreads();
    }
    if (threadIdx.x == 0) {
        float inv_n = 1.0f / (float)(CG_ * N_);
        float mean  = s_sum[0] * inv_n;
        float var   = s_sq[0] * inv_n - mean * mean;
        stats[2 * bg]     = mean;
        stats[2 * bg + 1] = rsqrtf(var + 1e-5f);
    }
}

// ---------------------------------------------------------------- weight f32 -> f16
__global__ void cvt_weights_kernel(const float* __restrict__ qw, const float* __restrict__ pw,
                                   _Float16* __restrict__ qw16, _Float16* __restrict__ pw16) {
    int i = blockIdx.x * 256 + threadIdx.x;
    if (i < C3_ * C_) qw16[i] = (_Float16)qw[i];
    if (i < C_ * C_)  pw16[i] = (_Float16)pw[i];
}

// ---------------------------------------------------------------- normalize + transpose (B,C,N)->(B,N,C) f16
__global__ void gn_apply_kernel(const float* __restrict__ x, const float* __restrict__ stats,
                                const float* __restrict__ w, const float* __restrict__ bch,
                                _Float16* __restrict__ h16) {
    __shared__ float tile[32][33];
    const int tx = threadIdx.x & 31, ty = threadIdx.x >> 5;   // 32 x 8
    const int n0 = blockIdx.x * 32;
    const int c0 = blockIdx.y * 32;
    const int b  = blockIdx.z;
    #pragma unroll
    for (int j = 0; j < 4; ++j) {
        int c = c0 + ty + j * 8;
        int g = c >> 4;
        float mean = stats[2 * (b * G_ + g)];
        float rsig = stats[2 * (b * G_ + g) + 1];
        float v = x[((size_t)b * C_ + c) * N_ + n0 + tx];     // coalesced along n
        tile[ty + j * 8][tx] = (v - mean) * rsig * w[c] + bch[c];
    }
    __syncthreads();
    #pragma unroll
    for (int j = 0; j < 4; ++j) {
        int n = n0 + ty + j * 8;
        int c = c0 + tx;
        h16[((size_t)b * N_ + n) * C_ + c] = (_Float16)tile[tx][ty + j * 8];  // coalesced along c
    }
}

// ---------------------------------------------------------------- WMMA fragment loaders
// A (16x32 f16, MxK): lane l -> row M=l&15, kBase=(l>>4)*16; halves 0..7 = K kBase..kBase+7,
//                     halves 8..15 = K kBase+8..kBase+15 (contiguous, matching B).
__device__ inline v16h load_a_frag(const _Float16* __restrict__ A, int lda,
                                   int row, int k0, int lane) {
    const _Float16* ap = A + (size_t)row * lda + k0 + ((lane >> 4) * 16);
    v8h lo = *(const v8h*)(ap);
    v8h hi = *(const v8h*)(ap + 8);
    return __builtin_shufflevector(lo, hi, 0,1,2,3,4,5,6,7, 8,9,10,11,12,13,14,15);
}
// B (32x16 f16, KxN) built from row-major W[m][k] (B = W^T tile): lane l -> col N=l&15,
// half j = K (l>>4)*16 + j  => 16 contiguous halves of W along k.
__device__ inline v16h load_b_frag(const _Float16* __restrict__ W, int ldw,
                                   int col, int k0, int lane) {
    return *(const v16h*)(W + (size_t)col * ldw + k0 + ((lane >> 4) * 16));
}

#define WMMA_F16(a, b, c) \
    __builtin_amdgcn_wmma_f32_16x16x32_f16(false, (a), false, (b), (short)0, (c), false, false)

// ---------------------------------------------------------------- QKV GEMM: (R,512)x(512,1536) -> f16
// Block = 8 waves as 4(n) x 2(m); wave tile 32x32 (4 accumulators); block tile 128(n) x 64(m).
__global__ void qkv_gemm_kernel(const _Float16* __restrict__ A,   // h16  [R][512]
                                const _Float16* __restrict__ W,   // qkvw [1536][512]
                                const float* __restrict__ bias,   // [1536]
                                _Float16* __restrict__ out) {     // [R][1536]
    const int lane = threadIdx.x & 31;
    const int wv   = threadIdx.x >> 5;
    const int n0   = blockIdx.x * 128 + (wv & 3) * 32;
    const int m0   = blockIdx.y * 64  + (wv >> 2) * 32;
    const int l15  = lane & 15;
    const int r0   = n0 + l15,  r1 = r0 + 16;   // A rows of the two row-subtiles
    const int c0   = m0 + l15,  c1 = c0 + 16;   // B cols of the two col-subtiles
    v8f acc00 = {}, acc01 = {}, acc10 = {}, acc11 = {};
    for (int k0 = 0; k0 < C_; k0 += 32) {
        v16h a0 = load_a_frag(A, C_, r0, k0, lane);
        v16h a1 = load_a_frag(A, C_, r1, k0, lane);
        v16h b0 = load_b_frag(W, C_, c0, k0, lane);
        v16h b1 = load_b_frag(W, C_, c1, k0, lane);
        acc00 = WMMA_F16(a0, b0, acc00);
        acc01 = WMMA_F16(a0, b1, acc01);
        acc10 = WMMA_F16(a1, b0, acc10);
        acc11 = WMMA_F16(a1, b1, acc11);
    }
    // D layout: VGPR r -> row M = r + (lane>>4)*8, col N = lane&15
    const int rb0 = n0 + (lane >> 4) * 8;       // rows of subtile 0
    const int rb1 = rb0 + 16;                   // rows of subtile 1
    const float bv0 = bias[c0], bv1 = bias[c1];
    #pragma unroll
    for (int r = 0; r < 8; ++r) {
        out[(size_t)(rb0 + r) * C3_ + c0] = (_Float16)(acc00[r] + bv0);
        out[(size_t)(rb0 + r) * C3_ + c1] = (_Float16)(acc01[r] + bv1);
        out[(size_t)(rb1 + r) * C3_ + c0] = (_Float16)(acc10[r] + bv0);
        out[(size_t)(rb1 + r) * C3_ + c1] = (_Float16)(acc11[r] + bv1);
    }
}

// ---------------------------------------------------------------- per-pixel 8x8 head attention
// qkv row layout: m = s*512 + head*64 + d, s in {q,k,v}. 256 threads = 4 pixels/block.
__global__ void attn_kernel(const _Float16* __restrict__ qkv, _Float16* __restrict__ o16) {
    __shared__ _Float16 sq[4][512], sk[4][512], sv[4][512];
    __shared__ float S[4][64], P[4][64];
    const int t  = threadIdx.x & 63;           // lane within pixel slice
    const int px = threadIdx.x >> 6;           // 0..3
    const int p  = blockIdx.x * 4 + px;
    const _Float16* base = qkv + (size_t)p * C3_;
    for (int i = t; i < 512; i += 64) {
        sq[px][i] = base[i]; sk[px][i] = base[512 + i]; sv[px][i] = base[1024 + i];
    }
    __syncthreads();
    const int h = t >> 3, g = t & 7;
    float s = 0.f;
    #pragma unroll 8
    for (int d = 0; d < 64; ++d) s += (float)sq[px][h * 64 + d] * (float)sk[px][g * 64 + d];
    S[px][t] = s * 0.125f;                     // hd^-0.5 = 1/8
    __syncthreads();
    float mx = -1e30f;
    #pragma unroll
    for (int j = 0; j < 8; ++j) mx = fmaxf(mx, S[px][h * 8 + j]);
    float sum = 0.f;
    #pragma unroll
    for (int j = 0; j < 8; ++j) sum += __expf(S[px][h * 8 + j] - mx);
    P[px][t] = __expf(S[px][t] - mx) / sum;
    __syncthreads();
    const int d0 = g * 8;                      // thread covers o[h][d0..d0+7]
    #pragma unroll
    for (int d = 0; d < 8; ++d) {
        float acc = 0.f;
        #pragma unroll
        for (int gg = 0; gg < 8; ++gg)
            acc += P[px][h * 8 + gg] * (float)sv[px][gg * 64 + d0 + d];
        o16[(size_t)p * C_ + h * 64 + d0 + d] = (_Float16)acc;
    }
}

// ---------------------------------------------------------------- proj GEMM + bias + residual + transpose back
__device__ inline void store_residual(const v8f& acc, int row0, int c,
                                      const float* __restrict__ bias,
                                      const float* __restrict__ x,
                                      float* __restrict__ out) {
    const int b  = row0 >> 12;                 // / 4096
    const int nn = row0 & (N_ - 1);
    const size_t base = ((size_t)b * C_ + c) * N_ + nn;   // contiguous 8 floats along n
    const float bv = bias[c];
    #pragma unroll
    for (int r = 0; r < 8; ++r)
        out[base + r] = x[base + r] + acc[r] + bv;
}

__global__ void proj_gemm_kernel(const _Float16* __restrict__ A,   // o16 [R][512]
                                 const _Float16* __restrict__ W,   // projw [512][512]
                                 const float* __restrict__ bias,   // [512]
                                 const float* __restrict__ x,      // [B][C][N] residual
                                 float* __restrict__ out) {        // [B][C][N]
    const int lane = threadIdx.x & 31;
    const int wv   = threadIdx.x >> 5;
    const int n0   = blockIdx.x * 128 + (wv & 3) * 32;
    const int m0   = blockIdx.y * 64  + (wv >> 2) * 32;
    const int l15  = lane & 15;
    const int r0   = n0 + l15,  r1 = r0 + 16;
    const int c0   = m0 + l15,  c1 = c0 + 16;
    v8f acc00 = {}, acc01 = {}, acc10 = {}, acc11 = {};
    for (int k0 = 0; k0 < C_; k0 += 32) {
        v16h a0 = load_a_frag(A, C_, r0, k0, lane);
        v16h a1 = load_a_frag(A, C_, r1, k0, lane);
        v16h b0 = load_b_frag(W, C_, c0, k0, lane);
        v16h b1 = load_b_frag(W, C_, c1, k0, lane);
        acc00 = WMMA_F16(a0, b0, acc00);
        acc01 = WMMA_F16(a0, b1, acc01);
        acc10 = WMMA_F16(a1, b0, acc10);
        acc11 = WMMA_F16(a1, b1, acc11);
    }
    const int rb0 = n0 + (lane >> 4) * 8;      // flat rows; 128-row tiles never cross b (4096%128==0)
    const int rb1 = rb0 + 16;
    store_residual(acc00, rb0, c0, bias, x, out);
    store_residual(acc01, rb0, c1, bias, x, out);
    store_residual(acc10, rb1, c0, bias, x, out);
    store_residual(acc11, rb1, c1, bias, x, out);
}

// ---------------------------------------------------------------- launcher
extern "C" void kernel_launch(void* const* d_in, const int* in_sizes, int n_in,
                              void* d_out, int out_size, void* d_ws, size_t ws_size,
                              hipStream_t stream) {
    const float* x      = (const float*)d_in[0];
    const float* norm_w = (const float*)d_in[1];
    const float* norm_b = (const float*)d_in[2];
    const float* qkv_w  = (const float*)d_in[3];
    const float* qkv_b  = (const float*)d_in[4];
    const float* proj_w = (const float*)d_in[5];
    const float* proj_b = (const float*)d_in[6];
    float* out = (float*)d_out;

    char* ws = (char*)d_ws;
    float*    stats = (float*)ws;        ws += 4096;
    _Float16* qw16  = (_Float16*)ws;     ws += (size_t)C3_ * C_ * 2;   // 1.5 MB
    _Float16* pw16  = (_Float16*)ws;     ws += (size_t)C_  * C_ * 2;   // 0.5 MB
    _Float16* h16   = (_Float16*)ws;     ws += (size_t)R_  * C_ * 2;   // 64 MB
    _Float16* qkv16 = (_Float16*)ws;                                    // 192 MB
    _Float16* o16   = h16;               // h16 is dead after qkv_gemm; reuse for o

    gn_stats_kernel<<<B_ * G_, 256, 0, stream>>>(x, stats);
    cvt_weights_kernel<<<(C3_ * C_ + 255) / 256, 256, 0, stream>>>(qkv_w, proj_w, qw16, pw16);

    dim3 gnorm(N_ / 32, C_ / 32, B_);
    gn_apply_kernel<<<gnorm, 256, 0, stream>>>(x, stats, norm_w, norm_b, h16);

    dim3 g1(R_ / 128, C3_ / 64);
    qkv_gemm_kernel<<<g1, 256, 0, stream>>>(h16, qw16, qkv_b, qkv16);

    attn_kernel<<<R_ / 4, 256, 0, stream>>>(qkv16, o16);

    dim3 g2(R_ / 128, C_ / 64);
    proj_gemm_kernel<<<g2, 256, 0, stream>>>(o16, pw16, proj_b, x, out);
}
